// TransformerXL_61409442398568
// MI455X (gfx1250) — matrix operand order repaired and checked by reading, not compiled, not run on hardware
//
#include <hip/hip_runtime.h>
#include <hip/hip_bf16.h>

// ---------------- problem constants ----------------
constexpr int T_   = 1024;
constexpr int B_   = 4;
constexpr int D_   = 512;
constexpr int H_   = 8;
constexpr int HD_  = 64;
constexpr int FF_  = 2048;
constexpr int S_   = 1536;   // T + MEM
constexpr int MEM_ = 512;
constexpr int L_   = 2;
#define EPS_ 1e-5f

// ---------------- WMMA types ----------------
typedef __attribute__((ext_vector_type(16))) __bf16 v16bf;
typedef __attribute__((ext_vector_type(8)))  float  v8f;

union Frag { v16bf v; uint4 q[2]; };

__device__ __forceinline__ unsigned short f2b(float f) {
  unsigned int u = __float_as_uint(f);
  u += 0x7fffu + ((u >> 16) & 1u);     // round to nearest even
  return (unsigned short)(u >> 16);
}

// ---------------- batched bf16 WMMA GEMM: C = A * B^T, f32 accumulate --------
// A: [M,K] bf16 row-major, B: [N,K] bf16 row-major (NT).
// M % 128 == 0, N % 64 == 0, K % 32 == 0, lda/ldb % 8 == 0.
// batching: z -> (b = z / batchH, h = z % batchH) with per-b/per-h strides.
__global__ __launch_bounds__(256) void k_gemm(
    const unsigned short* __restrict__ A, const unsigned short* __restrict__ Bm,
    float* __restrict__ C,
    int M, int N, int K, int lda, int ldb, int ldc, int batchH,
    long long sAb, long long sAh, long long sBb, long long sBh,
    long long sCb, long long sCh)
{
  int z  = blockIdx.z;
  int bb = z / batchH, hh = z % batchH;
  A  += (size_t)bb * sAb + (size_t)hh * sAh;
  Bm += (size_t)bb * sBb + (size_t)hh * sBh;
  C  += (size_t)bb * sCb + (size_t)hh * sCh;

  constexpr int BM = 128, BN = 64, BK = 32, LW = 40;   // 80B rows, 16B aligned
  __shared__ __align__(16) unsigned short sA[2][BM][LW];
  __shared__ __align__(16) unsigned short sB[2][BN][LW];

  const int tid  = threadIdx.x;
  const int lane = tid & 31, wave = tid >> 5;
  const int wm = wave & 3, wn = wave >> 2;            // 4 waves along M, 2 along N
  const int m0 = blockIdx.y * BM;
  const int n0 = blockIdx.x * BN;
  const int laneHi = (lane >> 4) & 1;
  const int lmod   = lane & 15;

  // staging maps: A = 512 chunks of 8 bf16 (2/thread), B = 256 chunks (1/thread)
  const int ar0 = (tid + 0)   >> 2, ak0 = ((tid + 0)   & 3) << 3;
  const int ar1 = (tid + 256) >> 2, ak1 = ((tid + 256) & 3) << 3;
  const int br  = tid >> 2,         bk  = (tid & 3) << 3;

  const unsigned short* gA0 = &A[(size_t)(m0 + ar0) * lda + ak0];
  const unsigned short* gA1 = &A[(size_t)(m0 + ar1) * lda + ak1];
  const unsigned short* gB  = &Bm[(size_t)(n0 + br) * ldb + bk];

  v8f acc[2][2];
#pragma unroll
  for (int a = 0; a < 2; ++a)
#pragma unroll
    for (int b = 0; b < 2; ++b)
#pragma unroll
      for (int r = 0; r < 8; ++r) acc[a][b][r] = 0.0f;

  // prologue: stage tile 0
  *(uint4*)&sA[0][ar0][ak0] = *(const uint4*)gA0;
  *(uint4*)&sA[0][ar1][ak1] = *(const uint4*)gA1;
  *(uint4*)&sB[0][br][bk]   = *(const uint4*)gB;
  __syncthreads();

  const int nk = K / BK;
  for (int ki = 0; ki < nk; ++ki) {
    const int cur = ki & 1;
    const bool have = (ki + 1 < nk);

    // issue next tile's global loads into registers (no wait yet)
    uint4 ra0, ra1, rb0;
    if (have) {
      const int k1 = (ki + 1) * BK;
      ra0 = *(const uint4*)(gA0 + k1);
      ra1 = *(const uint4*)(gA1 + k1);
      rb0 = *(const uint4*)(gB + k1);
    }

    // compute on current tile (hides the global-load latency above)
    Frag af[2], bf[2];
#pragma unroll
    for (int mi = 0; mi < 2; ++mi) {
      const unsigned short* rp = &sA[cur][wm * 32 + mi * 16 + lmod][0];
      af[mi].q[0] = *(const uint4*)(rp + (laneHi << 3));        // k = laneHi*8 ..
      af[mi].q[1] = *(const uint4*)(rp + 16 + (laneHi << 3));   // k = 16+laneHi*8 ..
    }
#pragma unroll
    for (int ni = 0; ni < 2; ++ni) {
      const unsigned short* rp = &sB[cur][wn * 32 + ni * 16 + lmod][0];
      bf[ni].q[0] = *(const uint4*)(rp + (laneHi << 3));        // k = laneHi*8 ..
      bf[ni].q[1] = *(const uint4*)(rp + 16 + (laneHi << 3));
    }
#pragma unroll
    for (int mi = 0; mi < 2; ++mi)
#pragma unroll
      for (int ni = 0; ni < 2; ++ni)
        acc[mi][ni] = __builtin_amdgcn_wmma_f32_16x16x32_bf16(
            false, af[mi].v, false, bf[ni].v,
            (short)0, acc[mi][ni], false, false);

    // store staged registers into the alternate buffer, then sync
    if (have) {
      *(uint4*)&sA[cur ^ 1][ar0][ak0] = ra0;
      *(uint4*)&sA[cur ^ 1][ar1][ak1] = ra1;
      *(uint4*)&sB[cur ^ 1][br][bk]   = rb0;
    }
    __syncthreads();
  }

  // store C (f32): lane<16 -> N=lane, M=r; lane>=16 -> N=lane-16, M=r+8
#pragma unroll
  for (int mi = 0; mi < 2; ++mi) {
#pragma unroll
    for (int ni = 0; ni < 2; ++ni) {
      int n = n0 + wn * 32 + ni * 16 + lmod;
#pragma unroll
      for (int r = 0; r < 8; ++r) {
        int m = m0 + wm * 32 + mi * 16 + r + laneHi * 8;
        C[(size_t)m * ldc + n] = acc[mi][ni][r];
      }
    }
  }
}

// ---------------- elementwise / reduction kernels ----------------
// generic f32 -> bf16 convert
__global__ __launch_bounds__(256) void k_cvt(const float* __restrict__ in,
                                             unsigned short* __restrict__ out) {
  size_t idx = (size_t)blockIdx.x * 256 + threadIdx.x;
  out[idx] = f2b(in[idx]);
}

__global__ __launch_bounds__(256) void k_sinusoid(const int* __restrict__ pos,
                                                  unsigned short* __restrict__ pe) {
  int idx = blockIdx.x * 256 + threadIdx.x;                 // over S * D/2
  if (idx >= S_ * (D_ / 2)) return;
  int p = idx / (D_ / 2), i = idx % (D_ / 2);
  float div = expf((2.0f * (float)i) * (-logf(10000.0f) / (float)D_));
  float a = (float)pos[p] * div;
  pe[(size_t)p * D_ + 2 * i]     = f2b(sinf(a));
  pe[(size_t)p * D_ + 2 * i + 1] = f2b(cosf(a));
}

__global__ __launch_bounds__(256) void k_xs(const float* __restrict__ x,
                                            float* __restrict__ xs) {
  size_t idx = (size_t)blockIdx.x * 256 + threadIdx.x;      // over T*B*D
  size_t t = idx / (B_ * D_), r = idx % (B_ * D_);
  size_t b = r / D_, d = r % D_;
  xs[idx] = x[(b * T_ + t) * D_ + d];
}

// cat (bf16) = concat(mems[l], prev) along time
__global__ __launch_bounds__(256) void k_cat(const float* __restrict__ memsL,
                                             const float* __restrict__ prev,
                                             unsigned short* __restrict__ cat) {
  size_t idx = (size_t)blockIdx.x * 256 + threadIdx.x;      // over S*B*D
  size_t s = idx / (B_ * D_), r = idx % (B_ * D_);
  float v = (s < MEM_) ? memsL[idx] : prev[(s - MEM_) * (size_t)(B_ * D_) + r];
  cat[idx] = f2b(v);
}

// qkv[S,B,3*512] -> bf16 qu/qv [B,H,T,64] (+u/v bias), kk [B,H,S,64], vvT [B,H,64,S]
__global__ __launch_bounds__(256) void k_reorg_qkv(
    const float* __restrict__ qkv,
    const float* __restrict__ ub, const float* __restrict__ vb,
    unsigned short* __restrict__ qu, unsigned short* __restrict__ qv,
    unsigned short* __restrict__ kk, unsigned short* __restrict__ vvT) {
  size_t idx = (size_t)blockIdx.x * 256 + threadIdx.x;      // over S*B*512
  size_t s = idx / (B_ * 512), r = idx % (B_ * 512);
  int b = (int)(r / 512), e = (int)(r % 512);
  int h = e >> 6, d = e & 63;
  size_t base = (s * B_ + b) * (size_t)(3 * 512);
  kk[(((size_t)b * H_ + h) * S_ + s) * 64 + d]  = f2b(qkv[base + 512 + e]);
  vvT[(((size_t)b * H_ + h) * 64 + d) * S_ + s] = f2b(qkv[base + 1024 + e]);
  if (s >= MEM_) {
    size_t t = s - MEM_;
    size_t qidx = (((size_t)b * H_ + h) * T_ + t) * 64 + d;
    float q = qkv[base + e];
    qu[qidx] = f2b(q + ub[e]);
    qv[qidx] = f2b(q + vb[e]);
  }
}

// r[S,512] -> bf16 [H,S,64]
__global__ __launch_bounds__(256) void k_reorg_r(const float* __restrict__ r,
                                                 unsigned short* __restrict__ rb) {
  size_t idx = (size_t)blockIdx.x * 256 + threadIdx.x;      // over S*512
  size_t s = idx / 512;
  int e = (int)(idx % 512);
  rb[((size_t)(e >> 6) * S_ + s) * 64 + (e & 63)] = f2b(r[idx]);
}

// fused rel-shift + mask + scale + softmax; per-b [H,T,S]; writes bf16 attn
__global__ __launch_bounds__(256) void k_softmax(const float* __restrict__ sc,
                                                 const float* __restrict__ pm,
                                                 unsigned short* __restrict__ attn) {
  int i = blockIdx.x, h = blockIdx.y, t = threadIdx.x;
  const float* row  = sc + ((size_t)h * T_ + i) * S_;
  const float* prow = pm + ((size_t)h * T_ + i) * S_;
  unsigned short* arow = attn + ((size_t)h * T_ + i) * S_;
  const int lim = i + MEM_;                 // valid j <= lim (mask: j > i+MEM)
  const int shift = T_ - 1 - i;             // rel_shift: pos[i, j + T-1-i]
  const float scale = 0.04419417382415922f; // 1/sqrt(D)
  __shared__ float red[256];

  float mx = -1e30f;
  for (int j = t; j < S_; j += 256)
    if (j <= lim) mx = fmaxf(mx, (row[j] + prow[j + shift]) * scale);
  red[t] = mx; __syncthreads();
  for (int o = 128; o > 0; o >>= 1) { if (t < o) red[t] = fmaxf(red[t], red[t + o]); __syncthreads(); }
  mx = red[0]; __syncthreads();

  float sum = 0.0f;
  for (int j = t; j < S_; j += 256)
    if (j <= lim) sum += expf((row[j] + prow[j + shift]) * scale - mx);
  red[t] = sum; __syncthreads();
  for (int o = 128; o > 0; o >>= 1) { if (t < o) red[t] += red[t + o]; __syncthreads(); }
  float inv = 1.0f / red[0];

  for (int j = t; j < S_; j += 256) {
    float e = 0.0f;
    if (j <= lim) e = expf((row[j] + prow[j + shift]) * scale - mx) * inv;
    arow[j] = f2b(e);
  }
}

// out = LN(a + b (+ cbias)) * gamma + beta ; optional bf16 mirror of out
__global__ __launch_bounds__(256) void k_add_ln(
    const float* __restrict__ a, const float* __restrict__ bx,
    const float* __restrict__ cbias,
    const float* __restrict__ gamma, const float* __restrict__ beta,
    float* __restrict__ out, unsigned short* __restrict__ outb) {
  int row = blockIdx.x, t = threadIdx.x;
  const float* pa = a  + (size_t)row * D_;
  const float* pb = bx + (size_t)row * D_;
  float v0 = pa[t]       + pb[t]       + (cbias ? cbias[t]       : 0.0f);
  float v1 = pa[t + 256] + pb[t + 256] + (cbias ? cbias[t + 256] : 0.0f);
  __shared__ float r1[256], r2[256];
  r1[t] = v0 + v1; r2[t] = v0 * v0 + v1 * v1;
  __syncthreads();
  for (int o = 128; o > 0; o >>= 1) {
    if (t < o) { r1[t] += r1[t + o]; r2[t] += r2[t + o]; }
    __syncthreads();
  }
  float mean = r1[0] / (float)D_;
  float var  = r2[0] / (float)D_ - mean * mean;
  float inv  = rsqrtf(var + EPS_);
  float o0 = (v0 - mean) * inv * gamma[t]       + beta[t];
  float o1 = (v1 - mean) * inv * gamma[t + 256] + beta[t + 256];
  float* po = out + (size_t)row * D_;
  po[t] = o0; po[t + 256] = o1;
  if (outb) {
    unsigned short* pq = outb + (size_t)row * D_;
    pq[t] = f2b(o0); pq[t + 256] = f2b(o1);
  }
}

// bf16 out = relu(f32 in + bias)
__global__ __launch_bounds__(256) void k_bias_relu(const float* __restrict__ x,
                                                   const float* __restrict__ b,
                                                   unsigned short* __restrict__ o) {
  size_t idx = (size_t)blockIdx.x * 256 + threadIdx.x;      // over T*B*FF
  int f = (int)(idx % FF_);
  o[idx] = f2b(fmaxf(0.0f, x[idx] + b[f]));
}

__global__ __launch_bounds__(256) void k_final_tr(const float* __restrict__ h2,
                                                  float* __restrict__ out) {
  size_t idx = (size_t)blockIdx.x * 256 + threadIdx.x;      // over B*T*D
  size_t b = idx / ((size_t)T_ * D_), r = idx % ((size_t)T_ * D_);
  size_t t = r / D_, d = r % D_;
  out[idx] = h2[(t * B_ + b) * D_ + d];
}

__global__ __launch_bounds__(256) void k_new_mems(
    const float* __restrict__ xs, const float* __restrict__ h1,
    const float* __restrict__ h2, float* __restrict__ out) {
  size_t idx = (size_t)blockIdx.x * 256 + threadIdx.x;      // over 3*MEM*B*D
  int i = (int)(idx / ((size_t)MEM_ * B_ * D_));
  size_t r = idx % ((size_t)MEM_ * B_ * D_);
  size_t m = r / (B_ * D_), c = r % (B_ * D_);
  const float* src = (i == 0) ? xs : (i == 1) ? h1 : h2;
  out[(size_t)B_ * T_ * D_ + idx] = src[((size_t)(T_ - MEM_) + m) * (B_ * D_) + c];
}

// ---------------- host orchestration ----------------
extern "C" void kernel_launch(void* const* d_in, const int* in_sizes, int n_in,
                              void* d_out, int out_size, void* d_ws, size_t ws_size,
                              hipStream_t stream) {
  (void)in_sizes; (void)n_in; (void)out_size; (void)ws_size;
  const float* x      = (const float*)d_in[0];
  const int*   posv   = (const int*)  d_in[1];
  /* d_in[2] = attn_mask (computed analytically) */
  const float* mems   = (const float*)d_in[3];
  const float* u_bias = (const float*)d_in[4];
  const float* v_bias = (const float*)d_in[5];
  const float* qkv_w  = (const float*)d_in[6];
  const float* pos_w  = (const float*)d_in[7];
  const float* out_w  = (const float*)d_in[8];
  const float* lin1_w = (const float*)d_in[9];
  const float* lin1_b = (const float*)d_in[10];
  const float* lin2_w = (const float*)d_in[11];
  const float* lin2_b = (const float*)d_in[12];
  const float* ln1_s  = (const float*)d_in[13];
  const float* ln1_b  = (const float*)d_in[14];
  const float* ln2_s  = (const float*)d_in[15];
  const float* ln2_b  = (const float*)d_in[16];
  float* out = (float*)d_out;

  // ---- workspace layout ----
  float* f = (float*)d_ws;
  float* xs   = f; f += (size_t)T_ * B_ * D_;
  float* h1   = f; f += (size_t)T_ * B_ * D_;
  float* h2   = f; f += (size_t)T_ * B_ * D_;
  float* qkvb = f; f += (size_t)S_ * B_ * 3 * 512;
  float* rbuf = f; f += (size_t)S_ * 512;
  float* ctx  = f; f += (size_t)T_ * B_ * 512;
  float* aout = f; f += (size_t)T_ * B_ * D_;
  float* hbuf = f; f += (size_t)T_ * B_ * D_;
  float* ff2  = f; f += (size_t)T_ * B_ * D_;
  float* sc   = f; f += (size_t)H_ * T_ * S_;   // per-b content scores
  float* pm   = f; f += (size_t)H_ * T_ * S_;   // per-b pos scores; reused as ff1
  float* ff1  = pm;                              // lifetimes disjoint (48MB >= 32MB)
  unsigned short* us = (unsigned short*)f;
  unsigned short* catb   = us; us += (size_t)S_ * B_ * D_;
  unsigned short* pebf   = us; us += (size_t)S_ * D_;
  unsigned short* qu     = us; us += (size_t)B_ * H_ * T_ * HD_;
  unsigned short* qv     = us; us += (size_t)B_ * H_ * T_ * HD_;
  unsigned short* kk     = us; us += (size_t)B_ * H_ * S_ * HD_;
  unsigned short* vvT    = us; us += (size_t)B_ * H_ * HD_ * S_;
  unsigned short* rb     = us; us += (size_t)H_ * S_ * HD_;
  unsigned short* attnb  = us; us += (size_t)H_ * T_ * S_;   // per-b bf16 attn
  unsigned short* ctxb   = us; us += (size_t)T_ * B_ * 512;
  unsigned short* hbufb  = us; us += (size_t)T_ * B_ * D_;
  unsigned short* ff1b   = us; us += (size_t)T_ * B_ * FF_;
  unsigned short* qkv_wb = us; us += (size_t)L_ * 3 * 512 * D_;
  unsigned short* pos_wb = us; us += (size_t)L_ * 512 * D_;
  unsigned short* out_wb = us; us += (size_t)L_ * D_ * 512;
  unsigned short* lin1_wb= us; us += (size_t)L_ * FF_ * D_;
  unsigned short* lin2_wb= us; us += (size_t)L_ * D_ * FF_;

  const int TB = T_ * B_;                       // 4096 rows
  const int SB = S_ * B_;                       // 6144 rows

  // one-time (per launch) weight conversion to bf16
  k_cvt<<<((size_t)L_ * 3 * 512 * D_) / 256, 256, 0, stream>>>(qkv_w,  qkv_wb);
  k_cvt<<<((size_t)L_ * 512 * D_)     / 256, 256, 0, stream>>>(pos_w,  pos_wb);
  k_cvt<<<((size_t)L_ * D_ * 512)     / 256, 256, 0, stream>>>(out_w,  out_wb);
  k_cvt<<<((size_t)L_ * FF_ * D_)     / 256, 256, 0, stream>>>(lin1_w, lin1_wb);
  k_cvt<<<((size_t)L_ * D_ * FF_)     / 256, 256, 0, stream>>>(lin2_w, lin2_wb);

  k_sinusoid<<<(S_ * (D_ / 2) + 255) / 256, 256, 0, stream>>>(posv, pebf);
  k_xs<<<((size_t)T_ * B_ * D_) / 256, 256, 0, stream>>>(x, xs);

  for (int l = 0; l < L_; ++l) {
    const float* prev = (l == 0) ? xs : h1;
    float* outp       = (l == 0) ? h1 : h2;

    k_cat<<<((size_t)S_ * B_ * D_) / 256, 256, 0, stream>>>(
        mems + (size_t)l * MEM_ * B_ * D_, prev, catb);

    // qkv = cat[SB,512] * qkv_w[l][1536,512]^T
    k_gemm<<<dim3(1536 / 64, SB / 128, 1), 256, 0, stream>>>(
        catb, qkv_wb + (size_t)l * 1536 * 512, qkvb,
        SB, 1536, 512, 512, 512, 1536, 1, 0, 0, 0, 0, 0, 0);

    // r = pe[S,512] * pos_w[l][512,512]^T
    k_gemm<<<dim3(512 / 64, S_ / 128, 1), 256, 0, stream>>>(
        pebf, pos_wb + (size_t)l * 512 * 512, rbuf,
        S_, 512, 512, 512, 512, 512, 1, 0, 0, 0, 0, 0, 0);

    k_reorg_qkv<<<((size_t)S_ * B_ * 512) / 256, 256, 0, stream>>>(
        qkvb, u_bias, v_bias, qu, qv, kk, vvT);
    k_reorg_r<<<((size_t)S_ * 512) / 256, 256, 0, stream>>>(rbuf, rb);

    for (int b = 0; b < B_; ++b) {
      const size_t qoff = (size_t)b * H_ * T_ * HD_;
      const size_t koff = (size_t)b * H_ * S_ * HD_;
      // content[h] = qu[h][T,64] * kk[h][S,64]^T
      k_gemm<<<dim3(S_ / 64, T_ / 128, H_), 256, 0, stream>>>(
          qu + qoff, kk + koff, sc,
          T_, S_, 64, 64, 64, S_, H_,
          0, (long long)T_ * 64, 0, (long long)S_ * 64, 0, (long long)T_ * S_);
      // pos[h] = qv[h][T,64] * rb[h][S,64]^T   (rb shared over b)
      k_gemm<<<dim3(S_ / 64, T_ / 128, H_), 256, 0, stream>>>(
          qv + qoff, rb, pm,
          T_, S_, 64, 64, 64, S_, H_,
          0, (long long)T_ * 64, 0, (long long)S_ * 64, 0, (long long)T_ * S_);

      k_softmax<<<dim3(T_, H_), 256, 0, stream>>>(sc, pm, attnb);

      // ctx[h] = attn[h][T,S] * vvT[h][64,S]^T -> ctx[t][b][h*64+d]
      k_gemm<<<dim3(1, T_ / 128, H_), 256, 0, stream>>>(
          attnb, vvT + koff, ctx + (size_t)b * 512,
          T_, 64, S_, S_, S_, B_ * 512, H_,
          0, (long long)T_ * S_, 0, (long long)64 * S_, 0, 64);
    }

    // a_out = ctx[TB,512] * out_w[l][512,512]^T
    k_cvt<<<((size_t)TB * 512) / 256, 256, 0, stream>>>(ctx, ctxb);
    k_gemm<<<dim3(512 / 64, TB / 128, 1), 256, 0, stream>>>(
        ctxb, out_wb + (size_t)l * 512 * 512, aout,
        TB, 512, 512, 512, 512, 512, 1, 0, 0, 0, 0, 0, 0);

    // h = LN(prev + a_out)  (+ bf16 mirror for FFN input)
    k_add_ln<<<TB, 256, 0, stream>>>(prev, aout, nullptr,
                                     ln1_s + (size_t)l * D_, ln1_b + (size_t)l * D_,
                                     hbuf, hbufb);

    // ff1 = h * lin1_w^T ; bf16 relu(+bias) ; ff2 = ff1 * lin2_w^T
    k_gemm<<<dim3(FF_ / 64, TB / 128, 1), 256, 0, stream>>>(
        hbufb, lin1_wb + (size_t)l * FF_ * D_, ff1,
        TB, FF_, 512, 512, 512, FF_, 1, 0, 0, 0, 0, 0, 0);
    k_bias_relu<<<((size_t)TB * FF_) / 256, 256, 0, stream>>>(
        ff1, lin1_b + (size_t)l * FF_, ff1b);
    k_gemm<<<dim3(512 / 64, TB / 128, 1), 256, 0, stream>>>(
        ff1b, lin2_wb + (size_t)l * D_ * FF_, ff2,
        TB, 512, FF_, FF_, FF_, 512, 1, 0, 0, 0, 0, 0, 0);

    // out = LN(h + ff2 + lin2_b)
    k_add_ln<<<TB, 256, 0, stream>>>(hbuf, ff2, lin2_b + (size_t)l * D_,
                                     ln2_s + (size_t)l * D_, ln2_b + (size_t)l * D_,
                                     outp, nullptr);
  }

  k_final_tr<<<((size_t)B_ * T_ * D_) / 256, 256, 0, stream>>>(h2, out);
  k_new_mems<<<((size_t)3 * MEM_ * B_ * D_) / 256, 256, 0, stream>>>(xs, h1, h2, out);
}
